// MyopicAttention_36361193128319
// MI455X (gfx1250) — matrix operand order repaired and checked by reading, not compiled, not run on hardware
//
#include <hip/hip_runtime.h>
#include <hip/hip_bf16.h>
#include <stdint.h>

// ---------------------------------------------------------------------------
// MyopicAttention for MI455X (gfx1250, wave32, WMMA + async global->LDS)
// Pipeline:
//   K0  f32 -> bf16 conversions (x, Wqkv, Wout, pb_w1)
//   K1  WMMA GEMM (bf16 x bf16 -> f32): qkv = x@Wqkv + b, scattered to q/k/v
//   K2a pos-bias layer0 (outer product + SiLU) -> bf16
//   K2b WMMA GEMM: h1 = silu(h0@W1 + b1) (f32)
//   K2c skinny GEMV: pbtab = h1@W2 + b2  (4095 x 8)
//   K3  threefry noise + score + 64-smallest selection per (b,h,block)
//   K4  gather + 4x64 attention core (VALU; M=4 too small for WMMA tiles)
//   K5  WMMA GEMM: out = attn_out@Wout + bout  (f32 -> d_out)
// A-tiles are staged with GLOBAL_LOAD_ASYNC_TO_LDS_B128 (ASYNCcnt path);
// B-tiles need a major-order change so they go through a VGPR transpose.
// ---------------------------------------------------------------------------

typedef __bf16 bf16;
typedef bf16  v16bf __attribute__((ext_vector_type(16)));
typedef bf16  v8bf  __attribute__((ext_vector_type(8)));
typedef float v8f   __attribute__((ext_vector_type(8)));

static constexpr int cB  = 4;
static constexpr int cN  = 2048;
static constexpr int cH  = 8;
static constexpr int cD  = 64;
static constexpr int cW  = 4;
static constexpr int cKP = 64;      // KEEP
static constexpr int cNW = cN / cW; // 512
static constexpr int cNF = 512;
static constexpr int cM  = cB * cN; // 8192 rows for the big GEMMs
static constexpr int cPB = 4096;    // padded 2N-1 = 4095 -> 4096

// ------------------------------ utilities ----------------------------------

__global__ void cvt_bf16_kernel(const float* __restrict__ in,
                                bf16* __restrict__ out, int n) {
  int i = blockIdx.x * blockDim.x + threadIdx.x;
  if (i < n) out[i] = (bf16)in[i];
}

// ------------------------------ WMMA GEMM ----------------------------------
// C[M,N] = A[M,K](bf16) @ B[K,N](bf16) + bias[N], fp32 accumulate.
// block = 256 threads (8 waves). Block tile 128(M) x 64(N); wave tile 32x32.
// mode 0: C0[m*N+n] = acc+bias
// mode 1: C0[m*N+n] = silu(acc+bias)
// mode 2: QKV scatter: col = h*192 + d*3 + c  ->  {C0,C1,C2}[((b*8+h)*2048+n)*64+d]
__global__ __launch_bounds__(256)
void wmma_gemm_kernel(const bf16* __restrict__ A, const bf16* __restrict__ Bm,
                      const float* __restrict__ bias,
                      int M, int N, int K,
                      float* __restrict__ C0, float* __restrict__ C1,
                      float* __restrict__ C2, int mode) {
  __shared__ __align__(16) bf16 lA[128 * 32];       // row-major [row][k]
  __shared__ __align__(16) bf16 lB[64 * 32];        // col-major [col][k]

  const int t    = threadIdx.x;
  const int lane = t & 31;
  const int wave = t >> 5;
  const int wm   = wave & 3;   // 4 row groups of 32
  const int wn   = wave >> 2;  // 2 col groups of 32
  const int m0   = blockIdx.y * 128;
  const int n0   = blockIdx.x * 64;

  v8f acc[2][2] = {};

  const int arow = t >> 1, ah = (t & 1) * 16;  // A DMA: 32 B per thread
  const int bk   = t >> 3, bc = (t & 7) * 8;   // B loader: 8 bf16 per thread

  const int r16 = lane & 15;
  const int hif = lane >> 4;
  const int kbA = hif * 8;    // A fragment K-phase per ISA layout
  const int kbB = hif * 8;    // B fragment K-phase per ISA layout

  // LDS byte offset of this thread's A-tile destination (flat addr[31:0]
  // is the LDS offset within the aperture).
  const uint32_t lA_dst = (uint32_t)(uintptr_t)(&lA[arow * 32 + ah]);

  for (int k0 = 0; k0 < K; k0 += 32) {
    // --- A tile 128x32: async global->LDS DMA, no VGPR round trip ---------
    {
      const uint32_t goff =
          (uint32_t)(((size_t)(m0 + arow) * (size_t)K + (size_t)(k0 + ah)) * 2u);
      asm volatile(
          "global_load_async_to_lds_b128 %0, %1, %2\n\t"
          "global_load_async_to_lds_b128 %0, %1, %2 offset:16"
          :
          : "v"(lA_dst), "v"(goff), "s"(A)
          : "memory");
    }
    // --- B tile 32x64: load + transpose to col-major in LDS ---------------
    v8bf br = *reinterpret_cast<const v8bf*>(Bm + (size_t)(k0 + bk) * N + n0 + bc);
#pragma unroll
    for (int e = 0; e < 8; ++e) lB[(bc + e) * 32 + bk] = br[e];
    // prefetch next B slab into L2 while we compute this one
    if (k0 + 32 < K)
      __builtin_prefetch(Bm + (size_t)(k0 + 32 + bk) * N + n0 + bc, 0, 3);

    asm volatile("s_wait_asynccnt 0x0" ::: "memory");  // A DMA landed in LDS
    __syncthreads();                                   // + B scatter visible

#pragma unroll
    for (int i = 0; i < 2; ++i) {
      const int row = wm * 32 + i * 16 + r16;
      union { v16bf v; v8bf h[2]; } af;
      af.h[0] = *reinterpret_cast<const v8bf*>(&lA[row * 32 + kbA]);
      af.h[1] = *reinterpret_cast<const v8bf*>(&lA[row * 32 + 16 + kbA]);
#pragma unroll
      for (int j = 0; j < 2; ++j) {
        const int col = wn * 32 + j * 16 + r16;
        union { v16bf v; v8bf h[2]; } bfg;
        bfg.h[0] = *reinterpret_cast<const v8bf*>(&lB[col * 32 + kbB]);
        bfg.h[1] = *reinterpret_cast<const v8bf*>(&lB[col * 32 + 16 + kbB]);
        acc[i][j] = __builtin_amdgcn_wmma_f32_16x16x32_bf16(
            false, af.v, false, bfg.v, (short)0, acc[i][j], false, false);
      }
    }
    __syncthreads();
  }

  // epilogue: C/D layout — element r: M = r + 8*(lane>=16), N = lane%16
#pragma unroll
  for (int i = 0; i < 2; ++i)
#pragma unroll
    for (int j = 0; j < 2; ++j)
#pragma unroll
      for (int r = 0; r < 8; ++r) {
        const int grow = m0 + wm * 32 + i * 16 + r + 8 * hif;
        const int gcol = n0 + wn * 32 + j * 16 + r16;
        float val = acc[i][j][r] + bias[gcol];
        if (mode == 0) {
          C0[(size_t)grow * N + gcol] = val;
        } else if (mode == 1) {
          C0[(size_t)grow * N + gcol] = val / (1.0f + expf(-val));  // SiLU
        } else {
          const int h  = gcol / 192;
          const int rr = gcol - h * 192;
          const int d  = rr / 3;
          const int c  = rr - d * 3;
          const int b  = grow >> 11;        // /2048
          const int n  = grow & 2047;
          const size_t off = ((size_t)((b * 8 + h) * 2048 + n)) * 64 + d;
          float* dst = (c == 0) ? C0 : (c == 1) ? C1 : C2;
          dst[off] = val;
        }
      }
}

// --------------------------- position-bias MLP -----------------------------

__global__ __launch_bounds__(256)
void pb_h0_kernel(const float* __restrict__ w0, const float* __restrict__ b0,
                  bf16* __restrict__ h0) {
  const int i  = blockIdx.x;                 // 0..4095 (4095 = pad row)
  const int ii = (i < 4095) ? i : 4094;
  const float pos = (float)(ii - 2047);      // arange(-(N-1), N)
  for (int j = threadIdx.x; j < cNF; j += 256) {
    float xv = pos * w0[j] + b0[j];
    float s  = xv / (1.0f + expf(-xv));      // SiLU
    h0[(size_t)i * cNF + j] = (bf16)s;
  }
}

__global__ __launch_bounds__(256)
void pb_h2_kernel(const float* __restrict__ h1, const float* __restrict__ w2,
                  const float* __restrict__ b2, float* __restrict__ pbtab) {
  __shared__ float red[8][256];
  const int i = blockIdx.x;   // 0..4094
  const int t = threadIdx.x;
  float p[8] = {0, 0, 0, 0, 0, 0, 0, 0};
  for (int kk = t; kk < cNF; kk += 256) {
    float hv = h1[(size_t)i * cNF + kk];
#pragma unroll
    for (int h = 0; h < 8; ++h) p[h] += hv * w2[kk * 8 + h];
  }
#pragma unroll
  for (int h = 0; h < 8; ++h) red[h][t] = p[h];
  __syncthreads();
  for (int s = 128; s > 0; s >>= 1) {
    if (t < s) {
#pragma unroll
      for (int h = 0; h < 8; ++h) red[h][t] += red[h][t + s];
    }
    __syncthreads();
  }
  if (t < 8) pbtab[(size_t)i * 8 + t] = red[t][0] + b2[t];
}

// ------------------------- threefry + top-k select -------------------------

__device__ __forceinline__ uint32_t rotl32(uint32_t x, int d) {
  return (x << d) | (x >> (32 - d));
}

// JAX threefry2x32: 20 rounds, key injection every 4.
__device__ __forceinline__ void threefry2x32(uint32_t k0, uint32_t k1,
                                             uint32_t x0, uint32_t x1,
                                             uint32_t& o0, uint32_t& o1) {
  const uint32_t k2 = k0 ^ k1 ^ 0x1BD11BDAu;
#define TF_R4(a, b, c, d)                                   \
  x0 += x1; x1 = rotl32(x1, a); x1 ^= x0;                   \
  x0 += x1; x1 = rotl32(x1, b); x1 ^= x0;                   \
  x0 += x1; x1 = rotl32(x1, c); x1 ^= x0;                   \
  x0 += x1; x1 = rotl32(x1, d); x1 ^= x0;
  x0 += k0; x1 += k1;
  TF_R4(13, 15, 26, 6)  x0 += k1; x1 += k2 + 1u;
  TF_R4(17, 29, 16, 24) x0 += k2; x1 += k0 + 2u;
  TF_R4(13, 15, 26, 6)  x0 += k0; x1 += k1 + 3u;
  TF_R4(17, 29, 16, 24) x0 += k1; x1 += k2 + 4u;
  TF_R4(13, 15, 26, 6)  x0 += k2; x1 += k0 + 5u;
#undef TF_R4
  o0 = x0; o1 = x1;
}

// One block per (b,h,blockq) row: build 2048 scores in LDS, pick 64 smallest
// (ties -> lower index, matching lax.top_k), then sort indices ascending.
__global__ __launch_bounds__(256)
void topk_kernel(int* __restrict__ idxout) {
  __shared__ float sc[2048];
  __shared__ float rv[256];
  __shared__ int   ri[256];
  __shared__ int   sel[64];
  const int bid = blockIdx.x;      // b*H*NW + h*NW + nw
  const int t   = threadIdx.x;
  const int nw  = bid & (cNW - 1);
  const uint32_t HALF = 16777216u; // (B*H*NW*N)/2

#pragma unroll
  for (int e = 0; e < 8; ++e) {
    const int j = t + e * 256;
    const uint32_t fi = (uint32_t)bid * 2048u + (uint32_t)j;
    const bool hi = (fi >= HALF);
    uint32_t x0 = hi ? (fi - HALF) : fi;
    uint32_t x1 = x0 + HALF;
    uint32_t o0, o1;
    threefry2x32(0u, 1u, x0, x1, o0, o1);   // jax.random.key(1) -> (0,1)
    const uint32_t bits = hi ? o1 : o0;
    const float u = __uint_as_float((bits >> 9) | 0x3f800000u) - 1.0f;
    const float noise = 3.0f * expf(-0.5f * log1pf(-u));  // 3*pareto(alpha=2)
    const float dist  = fabsf((float)(nw - (j >> 2)));
    sc[j] = dist - noise;
  }
  __syncthreads();

  for (int it = 0; it < cKP; ++it) {
    float bv = 3.4e38f;
    int   bi = 0x7fffffff;
#pragma unroll
    for (int e = 0; e < 8; ++e) {
      const int j = t + e * 256;   // increasing j: strict '<' keeps lowest idx
      const float vv = sc[j];
      if (vv < bv) { bv = vv; bi = j; }
    }
    rv[t] = bv; ri[t] = bi;
    __syncthreads();
    for (int s = 128; s > 0; s >>= 1) {
      if (t < s) {
        const float ov = rv[t + s];
        const int   oi = ri[t + s];
        if (ov < rv[t] || (ov == rv[t] && oi < ri[t])) { rv[t] = ov; ri[t] = oi; }
      }
      __syncthreads();
    }
    if (t == 0) { sel[it] = ri[0]; sc[ri[0]] = 3.4e38f; }
    __syncthreads();
  }

  if (t == 0) {  // jnp.sort(idx): ascending
    for (int a = 1; a < cKP; ++a) {
      const int key = sel[a];
      int p = a - 1;
      while (p >= 0 && sel[p] > key) { sel[p + 1] = sel[p]; --p; }
      sel[p + 1] = key;
    }
  }
  __syncthreads();
  if (t < cKP) idxout[(size_t)bid * cKP + t] = sel[t];
}

// --------------------------- attention core --------------------------------
// One block (64 threads) per (b,h,blockq): q is 4x64, K_sel/V_sel 64x64.
__global__ __launch_bounds__(64)
void attn_kernel(const float* __restrict__ q, const float* __restrict__ k,
                 const float* __restrict__ v, const int* __restrict__ idx,
                 const float* __restrict__ pbtab, bf16* __restrict__ aout) {
  __shared__ float qs[4][64];
  __shared__ float sarr[4][64];
  __shared__ float earr[4][64];
  __shared__ int   iarr[64];
  const int bid = blockIdx.x;
  const int t   = threadIdx.x;              // 0..63
  const int nw  = bid & (cNW - 1);
  const int h   = (bid >> 9) & 7;
  const int b   = bid >> 12;
  const size_t bh = (size_t)(b * 8 + h) * cN;

  const float* qbase = q + (bh + (size_t)nw * 4) * cD;
#pragma unroll
  for (int w = 0; w < 4; ++w) qs[w][t] = qbase[w * cD + t];
  const int my = idx[(size_t)bid * cKP + t];
  iarr[t] = my;
  __syncthreads();

  // dots: thread t owns key z=t; stream its K row (coalesced), q via LDS bcast
  const float* krow = k + (bh + my) * cD;
  float acc[4] = {0.f, 0.f, 0.f, 0.f};
  for (int d = 0; d < cD; ++d) {
    const float kv = krow[d];
#pragma unroll
    for (int w = 0; w < 4; ++w) acc[w] += qs[w][d] * kv;
  }
#pragma unroll
  for (int w = 0; w < 4; ++w) {
    const int rel = nw * 4 + w - my + (cN - 1);   // 0..4094
    acc[w] = acc[w] * 0.125f + pbtab[(size_t)rel * 8 + h];
    sarr[w][t] = acc[w];
  }
  __syncthreads();

  float e[4];
#pragma unroll
  for (int w = 0; w < 4; ++w) {
    float m = -3.4e38f;
    for (int z = 0; z < 64; ++z) m = fmaxf(m, sarr[w][z]);
    e[w] = expf(acc[w] - m);
    earr[w][t] = e[w];
  }
  __syncthreads();
  float a[4];
#pragma unroll
  for (int w = 0; w < 4; ++w) {
    float ssum = 0.f;
    for (int z = 0; z < 64; ++z) ssum += earr[w][z];
    a[w] = e[w] / ssum;
  }
  __syncthreads();
#pragma unroll
  for (int w = 0; w < 4; ++w) sarr[w][t] = a[w];   // attn weights
  __syncthreads();

  // out[w][d=t] = sum_z attn[w][z] * V[z][d] ; V rows read coalesced per z
  float o[4] = {0.f, 0.f, 0.f, 0.f};
  for (int z = 0; z < 64; ++z) {
    const float* vrow = v + (bh + iarr[z]) * cD;
    const float vv = vrow[t];
#pragma unroll
    for (int w = 0; w < 4; ++w) o[w] += sarr[w][z] * vv;
  }
#pragma unroll
  for (int w = 0; w < 4; ++w)
    aout[((size_t)(b * cN + nw * 4 + w)) * (cH * cD) + h * cD + t] = (bf16)o[w];
}

// ------------------------------ launcher -----------------------------------

extern "C" void kernel_launch(void* const* d_in, const int* in_sizes, int n_in,
                              void* d_out, int out_size, void* d_ws, size_t ws_size,
                              hipStream_t stream) {
  (void)in_sizes; (void)n_in; (void)out_size; (void)ws_size;
  const float* x    = (const float*)d_in[0];
  const float* Wqkv = (const float*)d_in[1];
  const float* bqkv = (const float*)d_in[2];
  const float* Wout = (const float*)d_in[3];
  const float* bout = (const float*)d_in[4];
  const float* pw0  = (const float*)d_in[5];
  const float* pb0  = (const float*)d_in[6];
  const float* pw1  = (const float*)d_in[7];
  const float* pb1  = (const float*)d_in[8];
  const float* pw2  = (const float*)d_in[9];
  const float* pb2  = (const float*)d_in[10];

  char* ws = (char*)d_ws;
  size_t off = 0;
  auto wsalloc = [&](size_t bytes) -> char* {
    char* p = ws + off;
    off += (bytes + 255) & ~(size_t)255;
    return p;
  };
  bf16*  xbf    = (bf16*) wsalloc((size_t)cM * cNF * 2);       // 8 MB
  bf16*  wqkvbf = (bf16*) wsalloc((size_t)cNF * 1536 * 2);     // 1.5 MB
  bf16*  woutbf = (bf16*) wsalloc((size_t)cNF * cNF * 2);      // 0.5 MB
  bf16*  w1bf   = (bf16*) wsalloc((size_t)cNF * cNF * 2);      // 0.5 MB
  float* qf     = (float*)wsalloc((size_t)cB * cH * cN * cD * 4);  // 16 MB
  float* kf     = (float*)wsalloc((size_t)cB * cH * cN * cD * 4);  // 16 MB
  float* vf     = (float*)wsalloc((size_t)cB * cH * cN * cD * 4);  // 16 MB
  bf16*  h0bf   = (bf16*) wsalloc((size_t)cPB * cNF * 2);      // 4 MB
  float* h1f    = (float*)wsalloc((size_t)cPB * cNF * 4);      // 8 MB
  float* pbtab  = (float*)wsalloc((size_t)cPB * cH * 4);       // 128 KB
  int*   idxbuf = (int*)  wsalloc((size_t)cB * cH * cNW * cKP * 4); // 4 MB
  bf16*  aobf   = (bf16*) wsalloc((size_t)cM * cNF * 2);       // 8 MB

  // K0: bf16 conversions
  {
    int n;
    n = cM * cNF;    cvt_bf16_kernel<<<(n + 255) / 256, 256, 0, stream>>>(x,    xbf,    n);
    n = cNF * 1536;  cvt_bf16_kernel<<<(n + 255) / 256, 256, 0, stream>>>(Wqkv, wqkvbf, n);
    n = cNF * cNF;   cvt_bf16_kernel<<<(n + 255) / 256, 256, 0, stream>>>(Wout, woutbf, n);
    n = cNF * cNF;   cvt_bf16_kernel<<<(n + 255) / 256, 256, 0, stream>>>(pw1,  w1bf,   n);
  }

  // K1: QKV projection, scattered directly to (B,H,N,D) q/k/v
  {
    dim3 g(1536 / 64, cM / 128);
    wmma_gemm_kernel<<<g, 256, 0, stream>>>(xbf, wqkvbf, bqkv, cM, 1536, cNF,
                                            qf, kf, vf, /*mode=*/2);
  }

  // K2: position-bias MLP
  pb_h0_kernel<<<cPB, 256, 0, stream>>>(pw0, pb0, h0bf);
  {
    dim3 g(cNF / 64, cPB / 128);
    wmma_gemm_kernel<<<g, 256, 0, stream>>>(h0bf, w1bf, pb1, cPB, cNF, cNF,
                                            h1f, nullptr, nullptr, /*mode=*/1);
  }
  pb_h2_kernel<<<2 * cN - 1, 256, 0, stream>>>(h1f, pw2, pb2, pbtab);

  // K3: noisy top-k key selection (exact JAX threefry reproduction)
  topk_kernel<<<cB * cH * cNW, 256, 0, stream>>>(idxbuf);

  // K4: sparse-attention core
  attn_kernel<<<cB * cH * cNW, 64, 0, stream>>>(qf, kf, vf, idxbuf, pbtab, aobf);

  // K5: output projection -> d_out (f32)
  {
    dim3 g(cNF / 64, cM / 128);
    wmma_gemm_kernel<<<g, 256, 0, stream>>>(aobf, woutbf, bout, cM, cNF, cNF,
                                            (float*)d_out, nullptr, nullptr, /*mode=*/0);
  }
}
